// ScaledDotProductAttention_45741401702591
// MI455X (gfx1250) — matrix operand order repaired and checked by reading, not compiled, not run on hardware
//
#include <hip/hip_runtime.h>

#define S_LEN 4096
#define B_SZ 4
#define D_IN 1024
#define D_HEAD 64
#define NROWS (B_SZ * S_LEN)   // 16384

typedef __attribute__((ext_vector_type(16))) __bf16 v16bf;
typedef __attribute__((ext_vector_type(8)))  __bf16 v8bf;
typedef __attribute__((ext_vector_type(8)))  float  v8f;

// A-fragment loader (16-bit A 16x32 layout): 16 contiguous bf16 at p.
__device__ __forceinline__ v16bf load_a_frag_bf16(const __bf16* p) {
  v8bf lo = *(const v8bf*)p;
  v8bf hi = *(const v8bf*)(p + 8);
  v16bf r;
#pragma unroll
  for (int e = 0; e < 8; ++e) { r[e] = lo[e]; r[e + 8] = hi[e]; }
  return r;
}

// ---------------------------------------------------------------------------
// Kernel 1: fused q/k/v projection.  x[16384,1024] f32  x  W[1024,64] (x3)
// -> qkv bf16 [3][16384][64] in workspace.
// 128 threads = 4 waves, each wave owns a 16-row M tile -> 64 rows per WG,
// 256 workgroups (one per WGP on a big CDNA5 part).
// ---------------------------------------------------------------------------
__global__ __launch_bounds__(128) void proj_kernel(
    const float* __restrict__ x,
    const float* __restrict__ Wq, const float* __restrict__ bq,
    const float* __restrict__ Wk, const float* __restrict__ bk,
    const float* __restrict__ Wv, const float* __restrict__ bv,
    __bf16* __restrict__ qkv) {
  // B tile staged N-major: ldsB[n_all][k'] , n_all in [0,192), k' in [0,32)
  __shared__ __align__(32) __bf16 ldsB[192 * 32];

  const int tid  = threadIdx.x;
  const int wave = tid >> 5;
  const int lane = tid & 31;
  const int rowBase = blockIdx.x * 64 + wave * 16;

  const int col  = lane & 15;
  const int half = lane >> 4;          // 0: lanes 0-15, 1: lanes 16-31
  const int arow = rowBase + col;      // A-fragment row for this lane
  const int kb   = half ? 16 : 0;      // A-fragment K base within 32-chunk

  v8f acc[12] = {};                    // [3 matrices][4 n-tiles] of 16x16 f32

  for (int k0 = 0; k0 < D_IN; k0 += 32) {
    __syncthreads();
    // Cooperative stage of W tiles (f32 -> bf16, transposed to N-major).
    for (int idx = tid; idx < 3 * 64 * 32; idx += 128) {
      const int m  = idx >> 11;          // which matrix
      const int rm = idx & 2047;
      const int n  = rm & 63;
      const int kk = rm >> 6;
      const float* W = (m == 0) ? Wq : (m == 1) ? Wk : Wv;
      ldsB[(m * 64 + n) * 32 + kk] = (__bf16)W[(size_t)(k0 + kk) * 64 + n];
    }
    __syncthreads();

    // A fragment: 16 rows x 32 K of x, converted to bf16 in registers.
    const float* xrow = x + (size_t)arow * D_IN + k0 + kb;
    v8f u0 = *(const v8f*)xrow;
    v8f u1 = *(const v8f*)(xrow + 8);
    // Prefetch next K-chunk of x into cache (global_prefetch_b8).
    __builtin_prefetch((const void*)(xrow + 32), 0, 0);
    v16bf a;
#pragma unroll
    for (int e = 0; e < 8; ++e) { a[e] = (__bf16)u0[e]; a[8 + e] = (__bf16)u1[e]; }

    // Preload B fragments, then WMMA burst; ds_loads overlap the matrix pipe.
    v16bf bf[12];
#pragma unroll
    for (int t = 0; t < 12; ++t) {
      const __bf16* bp = ldsB + (t * 16 + col) * 32 + (half ? 16 : 0);
      bf[t] = *(const v16bf*)bp;       // 16 contiguous K values for this column
    }
#pragma unroll
    for (int t = 0; t < 12; ++t)
      acc[t] = __builtin_amdgcn_wmma_f32_16x16x32_bf16(
          false, a, false, bf[t], (short)0, acc[t], false, false);
  }

  // Bias add + bf16 store.  C layout: vgpr r -> row (r + 8*half), lane&15 -> col.
#pragma unroll
  for (int m = 0; m < 3; ++m) {
    const float* bias = (m == 0) ? bq : (m == 1) ? bk : bv;
#pragma unroll
    for (int t = 0; t < 4; ++t) {
      const int n = t * 16 + col;
      const float bval = bias[n];
      v8f c = acc[m * 4 + t];
#pragma unroll
      for (int r = 0; r < 8; ++r) {
        const int row = rowBase + half * 8 + r;
        qkv[(size_t)m * NROWS * D_HEAD + (size_t)row * D_HEAD + n] =
            (__bf16)(c[r] + bval);
      }
    }
  }
}

// ---------------------------------------------------------------------------
// Kernel 2: causal flash attention.  One WG = (batch, 64 Q rows), 4 waves,
// each wave owns 16 Q rows; 256 workgroups total.  Key loop in 32-wide
// blocks with online softmax.  K tile staged via the async DMA path
// (GLOBAL_LOAD_ASYNC_TO_LDS, ASYNCcnt), V tile staged transposed.
// ---------------------------------------------------------------------------
__global__ __launch_bounds__(128) void flash_kernel(
    const __bf16* __restrict__ qkv, float* __restrict__ out) {
  const __bf16* Q = qkv;
  const __bf16* K = qkv + (size_t)NROWS * D_HEAD;
  const __bf16* V = qkv + 2 * (size_t)NROWS * D_HEAD;

  __shared__ __align__(32) __bf16 ldsK[32 * D_HEAD];    // [key][dim] async-staged
  __shared__ __align__(32) __bf16 ldsVt[D_HEAD * 32];   // [dim][key] transposed V
  __shared__ __align__(32) __bf16 ldsP[4 * 16 * 32];    // per-wave P staging

  const int tid  = threadIdx.x;
  const int wave = tid >> 5;
  const int lane = tid & 31;
  const int col  = lane & 15;
  const int half = lane >> 4;
  const int b    = blockIdx.y;
  const int m0   = blockIdx.x * 64;        // Q row block within batch
  const int r0   = m0 + wave * 16;         // this wave's Q row base

  // LDS byte offset of the K tile destination for this thread (32B each).
  const unsigned ldsKoff = (unsigned)(size_t)(&ldsK[0]) + (unsigned)tid * 32u;

  // Load Q fragments: 16 rows x 64 dims = 2 chunks of 16x32.
  const int arow = r0 + col;
  const int kb   = half ? 16 : 0;
  v16bf qf[2];
#pragma unroll
  for (int c = 0; c < 2; ++c)
    qf[c] = load_a_frag_bf16(Q + (size_t)(b * S_LEN + arow) * D_HEAD + c * 32 + kb);

  v8f acc[4] = {};
  float mstat[8], lstat[8];
#pragma unroll
  for (int r = 0; r < 8; ++r) { mstat[r] = -1e30f; lstat[r] = 0.0f; }

  const int nblocks = m0 / 32 + 2;         // causal frontier for this WG

  for (int jb = 0; jb < nblocks; ++jb) {
    const int kbase = jb * 32;

    __syncthreads();   // previous iteration's LDS reads complete

    // Async DMA: stage the 4KB K tile [32 keys x 64 dims] into LDS.
    // 128 threads x 32B = 4096B, two b128 transfers per lane, ASYNCcnt-tracked.
    {
      const unsigned long long gK =
          (unsigned long long)(size_t)(K + (size_t)(b * S_LEN + kbase) * D_HEAD) +
          (unsigned long long)tid * 32ull;
      asm volatile("global_load_async_to_lds_b128 %0, %1, off"
                   :: "v"(ldsKoff), "v"(gK) : "memory");
      asm volatile("global_load_async_to_lds_b128 %0, %1, off offset:16"
                   :: "v"(ldsKoff + 16u), "v"(gK) : "memory");
    }

    // Stage V tile transposed: ldsVt[d][kk] = V[kbase+kk][d]
    for (int idx = tid; idx < D_HEAD * 32; idx += 128) {
      const int d  = idx & 63;
      const int kk = idx >> 6;
      ldsVt[d * 32 + kk] = V[(size_t)(b * S_LEN + kbase + kk) * D_HEAD + d];
    }

    asm volatile("s_wait_asynccnt 0x0" ::: "memory");  // K tile landed in LDS
    __syncthreads();

    if (kbase <= r0 + 15) {                // block intersects this wave's causal region
      // scores: 16 rows x 32 keys = 2 C tiles, each accumulating over K=64.
      v8f sc[2];
#pragma unroll
      for (int t = 0; t < 2; ++t) {
        v8f s = {};
#pragma unroll
        for (int c = 0; c < 2; ++c) {
          const __bf16* kp =
              ldsK + (size_t)(t * 16 + col) * D_HEAD + c * 32 + (half ? 16 : 0);
          v16bf kf = *(const v16bf*)kp;            // contiguous head-dims
          s = __builtin_amdgcn_wmma_f32_16x16x32_bf16(
              false, qf[c], false, kf, (short)0, s, false, false);
        }
        sc[t] = s;
      }

      // Scale, causal mask, row max.
      float rowmax[8];
#pragma unroll
      for (int r = 0; r < 8; ++r) rowmax[r] = -1e30f;
#pragma unroll
      for (int t = 0; t < 2; ++t) {
        const int key = kbase + t * 16 + col;
#pragma unroll
        for (int r = 0; r < 8; ++r) {
          const int row = r0 + half * 8 + r;
          float v = sc[t][r] * 0.125f;             // 1/sqrt(64)
          v = (key <= row) ? v : -1e30f;
          sc[t][r] = v;
          rowmax[r] = fmaxf(rowmax[r], v);
        }
      }
      // Reduce row max across the 16-lane half (xor masks stay within half).
#pragma unroll
      for (int r = 0; r < 8; ++r)
#pragma unroll
        for (int msk = 1; msk < 16; msk <<= 1)
          rowmax[r] = fmaxf(rowmax[r], __shfl_xor(rowmax[r], msk, 32));

      // Online softmax update.
      float corr[8], rowsum[8];
#pragma unroll
      for (int r = 0; r < 8; ++r) {
        const float mn = fmaxf(mstat[r], rowmax[r]);
        corr[r] = __expf(mstat[r] - mn);
        mstat[r] = mn;
        rowsum[r] = 0.0f;
      }
#pragma unroll
      for (int t = 0; t < 2; ++t)
#pragma unroll
        for (int r = 0; r < 8; ++r) {
          const float p = __expf(sc[t][r] - mstat[r]);
          sc[t][r] = p;
          rowsum[r] += p;
        }
#pragma unroll
      for (int r = 0; r < 8; ++r)
#pragma unroll
        for (int msk = 1; msk < 16; msk <<= 1)
          rowsum[r] += __shfl_xor(rowsum[r], msk, 32);
#pragma unroll
      for (int r = 0; r < 8; ++r) lstat[r] = lstat[r] * corr[r] + rowsum[r];
#pragma unroll
      for (int t = 0; t < 4; ++t)
#pragma unroll
        for (int r = 0; r < 8; ++r) acc[t][r] *= corr[r];

      // Re-layout P (C layout -> A layout) through this wave's LDS tile.
      __bf16* pw = ldsP + wave * (16 * 32);
#pragma unroll
      for (int t = 0; t < 2; ++t)
#pragma unroll
        for (int r = 0; r < 8; ++r)
          pw[(half * 8 + r) * 32 + t * 16 + col] = (__bf16)sc[t][r];
      const v16bf pf = load_a_frag_bf16(pw + col * 32 + kb);

      // P @ V : 4 output N-tiles over the 32-key chunk.
#pragma unroll
      for (int t = 0; t < 4; ++t) {
        const __bf16* vp = ldsVt + (t * 16 + col) * 32 + (half ? 16 : 0);
        v16bf vf = *(const v16bf*)vp;
        acc[t] = __builtin_amdgcn_wmma_f32_16x16x32_bf16(
            false, pf, false, vf, (short)0, acc[t], false, false);
      }
    }
  }

  // Epilogue: out = acc / l  (f32).
#pragma unroll
  for (int t = 0; t < 4; ++t)
#pragma unroll
    for (int r = 0; r < 8; ++r) {
      const int row = r0 + half * 8 + r;
      out[(size_t)(b * S_LEN + row) * D_HEAD + t * 16 + col] =
          acc[t][r] / lstat[r];
    }
}

// ---------------------------------------------------------------------------
extern "C" void kernel_launch(void* const* d_in, const int* in_sizes, int n_in,
                              void* d_out, int out_size, void* d_ws, size_t ws_size,
                              hipStream_t stream) {
  const float* x  = (const float*)d_in[0];
  const float* Wq = (const float*)d_in[1];
  const float* bq = (const float*)d_in[2];
  const float* Wk = (const float*)d_in[3];
  const float* bk = (const float*)d_in[4];
  const float* Wv = (const float*)d_in[5];
  const float* bv = (const float*)d_in[6];
  float* out  = (float*)d_out;
  __bf16* qkv = (__bf16*)d_ws;   // 3 * 16384 * 64 bf16 = 6 MB

  proj_kernel<<<dim3(NROWS / 64), 128, 0, stream>>>(x, Wq, bq, Wk, bk, Wv, bv, qkv);
  flash_kernel<<<dim3(S_LEN / 64, B_SZ), 128, 0, stream>>>(qkv, out);
}
